// MHA_45079976739512
// MI455X (gfx1250) — matrix operand order repaired and checked by reading, not compiled, not run on hardware
//
#include <hip/hip_runtime.h>

typedef _Float16 f16;
typedef __attribute__((ext_vector_type(16))) _Float16 v16h;
typedef __attribute__((ext_vector_type(8)))  float    v8f;
typedef __attribute__((ext_vector_type(4)))  int      v4i;

union AV  { v16h h; uint4 u[2]; };
union PK8 { f16 h[8]; uint4 u; };

#define WMMA_F16(A, B, C) \
  __builtin_amdgcn_wmma_f32_16x16x32_f16(false, (A), false, (B), (short)0, (C), false, false)

constexpr int Bb = 8, S = 1024, D = 1024, H = 16, DH = 64;
constexpr int NT = S / 32;  // 32-key tiles per sequence

#if __has_builtin(__builtin_amdgcn_global_load_async_to_lds_b128) && \
    __has_builtin(__builtin_amdgcn_s_wait_asynccnt)
#define HAVE_ASYNC_LDS 1
#else
#define HAVE_ASYNC_LDS 0
#endif

#if HAVE_ASYNC_LDS
typedef __attribute__((address_space(1))) v4i* gp_t;  // global int4*
typedef __attribute__((address_space(3))) v4i* lp_t;  // LDS int4*
__device__ __forceinline__ void async_b128(const void* g, void* l) {
  // generic->AS1 via 64-bit int; generic->AS3 via low 32 bits (flat-aperture
  // rule: the low 32 bits give the LDS offset).
  __builtin_amdgcn_global_load_async_to_lds_b128(
      (gp_t)(uintptr_t)g, (lp_t)(uint32_t)(uintptr_t)l, 0, 0);
}
#endif

// ---------------------------------------------------------------------------
// Kernel 1: per-head QKV projection via v_wmma_f32_16x16x32_f16.
// Writes Q,K row-major f16 [bh][s][d] and V transposed f16 [bh][d][s].
// ---------------------------------------------------------------------------
__global__ __launch_bounds__(128) void qkv_proj(
    const float* __restrict__ x,
    const float* __restrict__ Wq, const float* __restrict__ bq,
    const float* __restrict__ Wk, const float* __restrict__ bk,
    const float* __restrict__ Wv, const float* __restrict__ bv,
    f16* __restrict__ Q, f16* __restrict__ K, f16* __restrict__ Vt) {
  const int lane = threadIdx.x;
  const int l16 = lane & 15, hi = lane >> 4;
  const int st = blockIdx.x * 4 + threadIdx.y;  // 16-row s tile
  const int h = blockIdx.y, b = blockIdx.z;
  const int bh = b * H + h;
  const int s0 = st * 16;

  // A operand: x tile 16x64, two K=32 chunks, interleaved-half layout.
  AV ax[2];
  const float* xrow = x + ((size_t)b * S + s0 + l16) * D + h * DH;
#pragma unroll
  for (int c = 0; c < 2; ++c) {
    const float* p = xrow + c * 32 + hi * 8;
#pragma unroll
    for (int e = 0; e < 8; ++e) ax[c].h[e] = (f16)p[e];
#pragma unroll
    for (int e = 0; e < 8; ++e) ax[c].h[8 + e] = (f16)p[16 + e];
  }

  const float* Wm[3] = {Wq, Wk, Wv};
  const float* bm[3] = {bq, bk, bv};

#pragma unroll
  for (int mtx = 0; mtx < 3; ++mtx) {
    const float* W = Wm[mtx] + (size_t)h * DH * DH;
    const float* bias = bm[mtx] + h * DH;
#pragma unroll
    for (int nt = 0; nt < 4; ++nt) {
      const int e = nt * 16 + l16;  // output column (per-head feature)
      const float be = bias[e];
      v8f acc;
#pragma unroll
      for (int r = 0; r < 8; ++r) acc[r] = be;  // bias broadcast down column
#pragma unroll
      for (int c = 0; c < 2; ++c) {
        // B operand: B[k=d][n=e] = W[e][d]; lane = column e, K = hi*8+0..7, 16+hi*8+0..7.
        AV bw;
        const float* wp = W + (size_t)e * DH + c * 32 + hi * 8;
#pragma unroll
        for (int k = 0; k < 8; ++k) { bw.h[k] = (f16)wp[k]; bw.h[8 + k] = (f16)wp[16 + k]; }
        acc = WMMA_F16(ax[c].h, bw.h, acc);
      }
      if (mtx < 2) {
        f16* dst = (mtx == 0 ? Q : K) + (size_t)bh * S * DH;
#pragma unroll
        for (int r = 0; r < 8; ++r) {
          const int m = r + hi * 8;  // C layout: VGPR r -> row r / r+8
          dst[(size_t)(s0 + m) * DH + e] = (f16)acc[r];
        }
      } else {
        PK8 pk;
#pragma unroll
        for (int r = 0; r < 8; ++r) pk.h[r] = (f16)acc[r];
        *reinterpret_cast<uint4*>(Vt + ((size_t)bh * DH + e) * S + s0 + hi * 8) = pk.u;
      }
    }
  }
}

// ---------------------------------------------------------------------------
// Kernel 2: FlashAttention with double-buffered ASYNC global->LDS staging of
// K/V tiles (CDNA5 GLOBAL_LOAD_ASYNC_TO_LDS_B128 + s_wait_asynccnt), shared
// by 4 waves.  Falls back to synchronous staging if builtins are absent.
// ---------------------------------------------------------------------------
__global__ __launch_bounds__(128) void flash_attn(
    const f16* __restrict__ Q, const f16* __restrict__ Kg,
    const f16* __restrict__ Vt, float* __restrict__ out) {
#if HAVE_ASYNC_LDS
  __shared__ __align__(16) f16 kbuf[2][32 * 64];   // [buf][key][d]    8KB
  __shared__ __align__(16) f16 vbuf[2][64 * 32];   // [buf][feat][key] 8KB
#else
  __shared__ __align__(16) f16 kbuf[1][32 * 64];
  __shared__ __align__(16) f16 vbuf[1][64 * 32];
#endif
  __shared__ __align__(16) f16 pbuf[4 * 16 * 32];  // per-wave P       4KB
  const int lane = threadIdx.x;
  const int wid = threadIdx.y;
  const int tid = wid * 32 + lane;  // 0..127
  const int l16 = lane & 15, hi = lane >> 4;
  const int st = blockIdx.x * 4 + wid;
  const int h = blockIdx.y, b = blockIdx.z;
  const int bh = b * H + h;
  const int s0 = st * 16;
  f16* myp = pbuf + wid * (16 * 32);

  const f16* qb = Q + (size_t)bh * S * DH;
  const f16* kb = Kg + (size_t)bh * S * DH;
  const f16* vb = Vt + (size_t)bh * DH * S;

  // Q A-operand: 16x64 as two K=32 chunks (interleaved-half layout).
  AV aq[2];
#pragma unroll
  for (int c = 0; c < 2; ++c) {
    const f16* p = qb + (size_t)(s0 + l16) * DH + c * 32 + hi * 8;
    aq[c].u[0] = *reinterpret_cast<const uint4*>(p);
    aq[c].u[1] = *reinterpret_cast<const uint4*>(p + 16);
  }

  // Per-thread staging slices: K is a linear 4KB copy; V copies 16-half runs.
  const int vf = tid >> 1, vo = (tid & 1) * 16;

  v8f o[4];
  float m[8], l[8];
#pragma unroll
  for (int t = 0; t < 4; ++t)
#pragma unroll
    for (int r = 0; r < 8; ++r) o[t][r] = 0.f;
#pragma unroll
  for (int r = 0; r < 8; ++r) { m[r] = -__builtin_inff(); l[r] = 0.f; }

  const float scale = 0.125f;  // 1/sqrt(64)

#if HAVE_ASYNC_LDS
  // Prologue: kick off tile 0 into buffer 0 (4 async b128 per thread).
  {
    const f16* ksrc = kb + (size_t)tid * 16;
    async_b128(ksrc, &kbuf[0][tid * 16]);
    async_b128(ksrc + 8, &kbuf[0][tid * 16 + 8]);
    const f16* vsrc = vb + (size_t)vf * S + vo;
    async_b128(vsrc, &vbuf[0][vf * 32 + vo]);
    async_b128(vsrc + 8, &vbuf[0][vf * 32 + vo + 8]);
  }
#endif

  for (int j = 0; j < NT; ++j) {
    const int k0 = j * 32;
#if HAVE_ASYNC_LDS
    const int cur = j & 1;
    __builtin_amdgcn_s_wait_asynccnt(0);  // my tile-j async loads landed
    __syncthreads();  // everyone's tile j visible; prev reads of nxt done
    if (j + 1 < NT) {  // overlap tile j+1 staging with tile j compute
      const f16* ksrc = kb + (size_t)(k0 + 32) * DH + tid * 16;
      async_b128(ksrc, &kbuf[cur ^ 1][tid * 16]);
      async_b128(ksrc + 8, &kbuf[cur ^ 1][tid * 16 + 8]);
      const f16* vsrc = vb + (size_t)vf * S + k0 + 32 + vo;
      async_b128(vsrc, &vbuf[cur ^ 1][vf * 32 + vo]);
      async_b128(vsrc + 8, &vbuf[cur ^ 1][vf * 32 + vo + 8]);
    }
    const f16* kt = kbuf[cur];
    const f16* vt = vbuf[cur];
#else
    // Synchronous fallback: global -> regs -> LDS with two barriers.
    const f16* ksrc = kb + (size_t)k0 * DH + tid * 16;
    uint4 kr0 = *reinterpret_cast<const uint4*>(ksrc);
    uint4 kr1 = *reinterpret_cast<const uint4*>(ksrc + 8);
    const f16* vsrc = vb + (size_t)vf * S + k0 + vo;
    uint4 vr0 = *reinterpret_cast<const uint4*>(vsrc);
    uint4 vr1 = *reinterpret_cast<const uint4*>(vsrc + 8);
    if (j + 1 < NT) {
      __builtin_prefetch(ksrc + 32 * DH, 0, 3);
      __builtin_prefetch(vsrc + 32, 0, 3);
    }
    __syncthreads();
    *reinterpret_cast<uint4*>(&kbuf[0][tid * 16]) = kr0;
    *reinterpret_cast<uint4*>(&kbuf[0][tid * 16 + 8]) = kr1;
    *reinterpret_cast<uint4*>(&vbuf[0][vf * 32 + vo]) = vr0;
    *reinterpret_cast<uint4*>(&vbuf[0][vf * 32 + vo + 8]) = vr1;
    __syncthreads();
    const f16* kt = kbuf[0];
    const f16* vt = vbuf[0];
#endif

    // ---- S = Q . K^T : two 16x16 score tiles, K-dim 64 = 2 chunks ----
    v8f sc[2];
#pragma unroll
    for (int kg = 0; kg < 2; ++kg) {
      v8f c;
#pragma unroll
      for (int r = 0; r < 8; ++r) c[r] = 0.f;
#pragma unroll
      for (int cc = 0; cc < 2; ++cc) {
        AV bk;  // B: column = key (lane), K = hi*8 + 0..7, 16 + hi*8 + 0..7 feats
        const f16* p = kt + (kg * 16 + l16) * DH + cc * 32 + hi * 8;
        bk.u[0] = *reinterpret_cast<const uint4*>(p);
        bk.u[1] = *reinterpret_cast<const uint4*>(p + 16);
        c = WMMA_F16(aq[cc].h, bk.h, c);
      }
      sc[kg] = c;
    }

    // ---- online softmax (row r + hi*8 lives in VGPR r of this lane) ----
    float corr[8];
#pragma unroll
    for (int r = 0; r < 8; ++r) {
      float a0 = sc[0][r] * scale, a1 = sc[1][r] * scale;
      float mx = fmaxf(a0, a1);
#pragma unroll
      for (int w = 1; w < 16; w <<= 1) mx = fmaxf(mx, __shfl_xor(mx, w, 32));
      const float mn = fmaxf(m[r], mx);
      const float p0 = __expf(a0 - mn), p1 = __expf(a1 - mn);
      float rs = p0 + p1;
#pragma unroll
      for (int w = 1; w < 16; w <<= 1) rs += __shfl_xor(rs, w, 32);
      corr[r] = __expf(m[r] - mn);
      l[r] = l[r] * corr[r] + rs;
      m[r] = mn;
      sc[0][r] = p0;
      sc[1][r] = p1;
    }
#pragma unroll
    for (int t = 0; t < 4; ++t)
#pragma unroll
      for (int r = 0; r < 8; ++r) o[t][r] *= corr[r];

    // ---- C-layout -> A-layout transpose of P through LDS (f32 -> f16) ----
#pragma unroll
    for (int r = 0; r < 8; ++r) {
      const int row = r + hi * 8;
      myp[row * 32 + l16] = (f16)sc[0][r];
      myp[row * 32 + 16 + l16] = (f16)sc[1][r];
    }
    __syncthreads();  // P visible (cross-lane within wave, + wave sync)
    AV ap;
    const f16* pp = myp + l16 * 32 + hi * 8;
    ap.u[0] = *reinterpret_cast<const uint4*>(pp);
    ap.u[1] = *reinterpret_cast<const uint4*>(pp + 16);

    // ---- O += P . V from staged V^T tile ----
#pragma unroll
    for (int t = 0; t < 4; ++t) {
      AV bv;
      const f16* p = vt + (t * 16 + l16) * 32 + hi * 8;
      bv.u[0] = *reinterpret_cast<const uint4*>(p);
      bv.u[1] = *reinterpret_cast<const uint4*>(p + 16);
      o[t] = WMMA_F16(ap.h, bv.h, o[t]);
    }
  }

  // ---- epilogue: O / l, write f32 [B,S,D] ----
#pragma unroll
  for (int r = 0; r < 8; ++r) l[r] = 1.0f / l[r];
#pragma unroll
  for (int t = 0; t < 4; ++t) {
    const int e = h * DH + t * 16 + l16;
#pragma unroll
    for (int r = 0; r < 8; ++r) {
      const int mr = r + hi * 8;
      out[((size_t)b * S + s0 + mr) * D + e] = o[t][r] * l[r];
    }
  }
}

// ---------------------------------------------------------------------------
extern "C" void kernel_launch(void* const* d_in, const int* in_sizes, int n_in,
                              void* d_out, int out_size, void* d_ws, size_t ws_size,
                              hipStream_t stream) {
  (void)in_sizes; (void)n_in; (void)out_size; (void)ws_size;
  const float* x  = (const float*)d_in[0];
  const float* Wq = (const float*)d_in[1];
  const float* bq = (const float*)d_in[2];
  const float* Wk = (const float*)d_in[3];
  const float* bk = (const float*)d_in[4];
  const float* Wv = (const float*)d_in[5];
  const float* bv = (const float*)d_in[6];

  const size_t per = (size_t)Bb * H * S * DH;  // elements per f16 tensor
  f16* Qw = (f16*)d_ws;
  f16* Kw = Qw + per;
  f16* Vw = Kw + per;  // 48 MB total workspace

  dim3 blk(32, 4);
  dim3 grid(S / 16 / 4, H, Bb);
  qkv_proj<<<grid, blk, 0, stream>>>(x, Wq, bq, Wk, bk, Wv, bv, Qw, Kw, Vw);
  flash_attn<<<grid, blk, 0, stream>>>(Qw, Kw, Vw, (float*)d_out);
}
